// MoELayer_52166672777640
// MI455X (gfx1250) — matrix operand order repaired and checked by reading, not compiled, not run on hardware
//
#include <hip/hip_runtime.h>

// ---------------------------------------------------------------------------
// CDNA5 (gfx1250) MoE layer.
//   router -> per expert: [W1 transpose-convert] GEMM1(relu) ->
//             [W2 transpose-convert] GEMM2(route-weighted accumulate)
// GEMM operands pre-converted to bf16 (A [M][K], B^T [N][K]); the WMMA
// K-loop stages tiles with GLOBAL_LOAD_ASYNC_TO_LDS_B128 (ASYNCcnt) and
// computes with v_wmma_f32_16x16x32_bf16 (wave32).
// ---------------------------------------------------------------------------

typedef __attribute__((ext_vector_type(16))) __bf16 v16bf;
typedef __attribute__((ext_vector_type(8)))  float  v8f;

struct alignas(16) U4 { unsigned int a, b, c, d; };
struct U8 { U4 lo, hi; };
struct alignas(8) BF4 { __bf16 x, y, z, w; };

static __device__ __forceinline__ v16bf load_bf16_frag(const __bf16* p0, const __bf16* p1) {
    U8 u;
    u.lo = *(const U4*)p0;
    u.hi = *(const U4*)p1;
    return __builtin_bit_cast(v16bf, u);
}

// Async global->LDS 16B copy (CDNA5 GLOBAL_LOAD_ASYNC_TO_LDS_B128, ASYNCcnt).
// lds_off: byte offset within workgroup LDS (low 32 bits of generic pointer).
static __device__ __forceinline__ void async_copy_b128(void* lds_dst, const void* gsrc) {
    const unsigned loff = (unsigned)(unsigned long long)(uintptr_t)lds_dst;
    const unsigned long long gaddr = (unsigned long long)(uintptr_t)gsrc;
    asm volatile("global_load_async_to_lds_b128 %0, %1, off"
                 :: "v"(loff), "v"(gaddr)
                 : "memory");
}
static __device__ __forceinline__ void wait_async0() {
    asm volatile("s_wait_asynccnt 0" ::: "memory");
}

// ---------------------------------------------------------------------------
// Router: logits = x @ Wg -> softmax -> top-2 -> dense [B,8] weights.
// One wave (32 lanes) per token.
// ---------------------------------------------------------------------------
__global__ __launch_bounds__(256) void moe_router(const float* __restrict__ x,
                                                  const float* __restrict__ Wg,
                                                  float* __restrict__ wgt,
                                                  int D) {
    constexpr int E = 8;
    const int lane  = threadIdx.x & 31;
    const int wave  = threadIdx.x >> 5;
    const int token = blockIdx.x * 8 + wave;

    const float* xr = x + (size_t)token * D;
    float acc[E];
#pragma unroll
    for (int e = 0; e < E; ++e) acc[e] = 0.f;

    for (int d = lane; d < D; d += 32) {
        const float xv = xr[d];
        const float* wr = Wg + (size_t)d * E;
#pragma unroll
        for (int e = 0; e < E; ++e) acc[e] += xv * wr[e];
    }
#pragma unroll
    for (int off = 16; off > 0; off >>= 1) {
#pragma unroll
        for (int e = 0; e < E; ++e) acc[e] += __shfl_xor(acc[e], off, 32);
    }

    float mx = acc[0];
#pragma unroll
    for (int e = 1; e < E; ++e) mx = fmaxf(mx, acc[e]);
    float p[E], s = 0.f;
#pragma unroll
    for (int e = 0; e < E; ++e) { p[e] = __expf(acc[e] - mx); s += p[e]; }
    const float inv = 1.f / s;

    int i1 = 0;
#pragma unroll
    for (int e = 1; e < E; ++e) if (p[e] > p[i1]) i1 = e;
    int i2 = (i1 == 0) ? 1 : 0;
#pragma unroll
    for (int e = 0; e < E; ++e) if (e != i1 && p[e] > p[i2]) i2 = e;

    if (lane < E) {
        const float w = (lane == i1 || lane == i2) ? p[lane] * inv : 0.f;
        wgt[(size_t)token * E + lane] = w;
    }
}

// ---------------------------------------------------------------------------
// cvt_x: f32 [B][D] -> bf16 [B][D]; one float4 per thread.
// ---------------------------------------------------------------------------
__global__ __launch_bounds__(256) void moe_cvt_x(const float* __restrict__ src,
                                                 __bf16* __restrict__ dst) {
    const size_t i4 = ((size_t)blockIdx.x * 256 + threadIdx.x) * 4;
    const float4 f = *(const float4*)(src + i4);
    *(BF4*)(dst + i4) = BF4{(__bf16)f.x, (__bf16)f.y, (__bf16)f.z, (__bf16)f.w};
}

// ---------------------------------------------------------------------------
// cvt_wT: f32 W[K][N] -> bf16 WT[N][K] via 64x64 LDS tile transpose.
// ---------------------------------------------------------------------------
__global__ __launch_bounds__(256) void moe_cvt_wT(const float* __restrict__ W,
                                                  __bf16* __restrict__ WT,
                                                  int K, int N) {
    constexpr int LS = 72;              // 64 + 8 pad, keeps 16B alignment
    __shared__ __bf16 lds[64 * LS];

    const int tid = threadIdx.x;
    const int k0 = blockIdx.x * 64;
    const int n0 = blockIdx.y * 64;

#pragma unroll
    for (int i = 0; i < 4; ++i) {
        const int c  = tid + 256 * i;
        const int kk = c >> 4;          // 0..63
        const int nc = (c & 15) * 4;    // 0..60
        const float4 f = *(const float4*)(W + (size_t)(k0 + kk) * N + n0 + nc);
        lds[(nc + 0) * LS + kk] = (__bf16)f.x;
        lds[(nc + 1) * LS + kk] = (__bf16)f.y;
        lds[(nc + 2) * LS + kk] = (__bf16)f.z;
        lds[(nc + 3) * LS + kk] = (__bf16)f.w;
    }
    __syncthreads();
#pragma unroll
    for (int i = 0; i < 2; ++i) {
        const int c  = tid + 256 * i;
        const int n  = c >> 3;          // 0..63
        const int kc = (c & 7) * 8;     // 0..56
        *(U4*)(WT + (size_t)(n0 + n) * K + k0 + kc) = *(const U4*)&lds[n * LS + kc];
    }
}

// ---------------------------------------------------------------------------
// GEMM tiling: block tile 128x128, K-step 32 (one WMMA K).
// 8 waves: 2 (M) x 4 (N); each wave owns 64x32 = 4(M) x 2(N) WMMA tiles
// -> 8 v_wmma per K-step per wave. Double-buffered LDS; staging is
// GLOBAL_LOAD_ASYNC_TO_LDS_B128 (no VGPR round trip), 1 barrier/K-step.
// LDS row stride 40 elements (80 B) keeps fragment loads 16B-aligned.
// ---------------------------------------------------------------------------
constexpr int BM = 128, BN = 128, SA = 40, SB = 40;

// GEMM1: h = relu(xb @ W1T^T + b1e), h stored as bf16.
__global__ __launch_bounds__(256) void moe_gemm1_relu(const __bf16* __restrict__ A,
                                                      const __bf16* __restrict__ BT,
                                                      const float* __restrict__ bias,
                                                      __bf16* __restrict__ Hout,
                                                      int K, int N) {
    __shared__ __bf16 lA[2][BM * SA];
    __shared__ __bf16 lB[2][BN * SB];

    const int tid  = threadIdx.x;
    const int lane = tid & 31;
    const int wave = tid >> 5;
    const int wm   = wave >> 2;   // 0..1 (M)
    const int wn   = wave & 3;    // 0..3 (N)
    const int row0 = blockIdx.y * BM;
    const int col0 = blockIdx.x * BN;

    const int l15 = lane & 15;
    const int kh  = lane & 16;          // B frag K half: 0 or 16
    const int kb  = (lane & 16) >> 1;   // C-row offset +0/+8

    // Staging: 128x32 bf16 tiles in 16B chunks, 2 per thread per operand.
    const int trow[2] = { (tid + 0) >> 2, (tid + 256) >> 2 };
    const int tcol    = (tid & 3) * 8;

    v8f acc[4][2];
#pragma unroll
    for (int m = 0; m < 4; ++m)
#pragma unroll
        for (int n = 0; n < 2; ++n)
            acc[m][n] = (v8f){0.f, 0.f, 0.f, 0.f, 0.f, 0.f, 0.f, 0.f};

    const int nsteps = K >> 5;

    // Prologue: async-stage K-tile 0 into buffer 0.
#pragma unroll
    for (int i = 0; i < 2; ++i) {
        async_copy_b128(&lA[0][trow[i] * SA + tcol], A  + (size_t)(row0 + trow[i]) * K + tcol);
        async_copy_b128(&lB[0][trow[i] * SB + tcol], BT + (size_t)(col0 + trow[i]) * K + tcol);
    }
    wait_async0();
    __syncthreads();

    for (int s = 0; s < nsteps; ++s) {
        const int cur  = s & 1;
        const bool more = (s + 1) < nsteps;

        if (more) {
            const int kk  = (s + 1) << 5;
            const int nxt = cur ^ 1;
#pragma unroll
            for (int i = 0; i < 2; ++i) {
                async_copy_b128(&lA[nxt][trow[i] * SA + tcol],
                                A  + (size_t)(row0 + trow[i]) * K + kk + tcol);
                async_copy_b128(&lB[nxt][trow[i] * SB + tcol],
                                BT + (size_t)(col0 + trow[i]) * K + kk + tcol);
            }
            if (s + 2 < nsteps) {   // pull the K+64 weight tile toward L2
                const int kp = (s + 2) << 5;
                __builtin_prefetch(BT + (size_t)(col0 + trow[0]) * K + kp + tcol, 0, 1);
            }
        }

        // Compute on buffer `cur` (async-staged + barriered last iteration).
        const __bf16* bp0 = &lB[cur][(wn * 32 + l15) * SB + kh];
        const __bf16* bp1 = bp0 + 16 * SB;
        const v16bf b0 = load_bf16_frag(bp0, bp0 + 8);
        const v16bf b1 = load_bf16_frag(bp1, bp1 + 8);
#pragma unroll
        for (int m = 0; m < 4; ++m) {
            const __bf16* ap = &lA[cur][(wm * 64 + m * 16 + l15) * SA + kh];
            const v16bf a = load_bf16_frag(ap, ap + 8);
            acc[m][0] = __builtin_amdgcn_wmma_f32_16x16x32_bf16(
                false, a, false, b0, (short)0, acc[m][0], false, false);
            acc[m][1] = __builtin_amdgcn_wmma_f32_16x16x32_bf16(
                false, a, false, b1, (short)0, acc[m][1], false, false);
        }

        if (more) {
            wait_async0();
            __syncthreads();
        }
    }

    const int gc0 = col0 + wn * 32 + l15;
    const int gc1 = gc0 + 16;
    const float bv0 = bias[gc0];
    const float bv1 = bias[gc1];
#pragma unroll
    for (int m = 0; m < 4; ++m) {
#pragma unroll
        for (int r = 0; r < 8; ++r) {
            const int gr = row0 + wm * 64 + m * 16 + r + kb;
            Hout[(size_t)gr * N + gc0] = (__bf16)fmaxf(acc[m][0][r] + bv0, 0.f);
            Hout[(size_t)gr * N + gc1] = (__bf16)fmaxf(acc[m][1][r] + bv1, 0.f);
        }
    }
}

// GEMM2: out (+)= wgt[:,e] * (h @ W2T^T + b2e)
__global__ __launch_bounds__(256) void moe_gemm2_scale(const __bf16* __restrict__ A,
                                                       const __bf16* __restrict__ BT,
                                                       const float* __restrict__ bias,
                                                       const float* __restrict__ wgt,
                                                       float* __restrict__ out,
                                                       int K, int N,
                                                       int expert, int accumulate) {
    __shared__ __bf16 lA[2][BM * SA];
    __shared__ __bf16 lB[2][BN * SB];

    const int tid  = threadIdx.x;
    const int lane = tid & 31;
    const int wave = tid >> 5;
    const int wm   = wave >> 2;
    const int wn   = wave & 3;
    const int row0 = blockIdx.y * BM;
    const int col0 = blockIdx.x * BN;

    const int l15 = lane & 15;
    const int kh  = lane & 16;
    const int kb  = (lane & 16) >> 1;

    const int trow[2] = { (tid + 0) >> 2, (tid + 256) >> 2 };
    const int tcol    = (tid & 3) * 8;

    v8f acc[4][2];
#pragma unroll
    for (int m = 0; m < 4; ++m)
#pragma unroll
        for (int n = 0; n < 2; ++n)
            acc[m][n] = (v8f){0.f, 0.f, 0.f, 0.f, 0.f, 0.f, 0.f, 0.f};

    const int nsteps = K >> 5;

#pragma unroll
    for (int i = 0; i < 2; ++i) {
        async_copy_b128(&lA[0][trow[i] * SA + tcol], A  + (size_t)(row0 + trow[i]) * K + tcol);
        async_copy_b128(&lB[0][trow[i] * SB + tcol], BT + (size_t)(col0 + trow[i]) * K + tcol);
    }
    wait_async0();
    __syncthreads();

    for (int s = 0; s < nsteps; ++s) {
        const int cur  = s & 1;
        const bool more = (s + 1) < nsteps;

        if (more) {
            const int kk  = (s + 1) << 5;
            const int nxt = cur ^ 1;
#pragma unroll
            for (int i = 0; i < 2; ++i) {
                async_copy_b128(&lA[nxt][trow[i] * SA + tcol],
                                A  + (size_t)(row0 + trow[i]) * K + kk + tcol);
                async_copy_b128(&lB[nxt][trow[i] * SB + tcol],
                                BT + (size_t)(col0 + trow[i]) * K + kk + tcol);
            }
            if (s + 2 < nsteps) {
                const int kp = (s + 2) << 5;
                __builtin_prefetch(BT + (size_t)(col0 + trow[0]) * K + kp + tcol, 0, 1);
            }
        }

        const __bf16* bp0 = &lB[cur][(wn * 32 + l15) * SB + kh];
        const __bf16* bp1 = bp0 + 16 * SB;
        const v16bf b0 = load_bf16_frag(bp0, bp0 + 8);
        const v16bf b1 = load_bf16_frag(bp1, bp1 + 8);
#pragma unroll
        for (int m = 0; m < 4; ++m) {
            const __bf16* ap = &lA[cur][(wm * 64 + m * 16 + l15) * SA + kh];
            const v16bf a = load_bf16_frag(ap, ap + 8);
            acc[m][0] = __builtin_amdgcn_wmma_f32_16x16x32_bf16(
                false, a, false, b0, (short)0, acc[m][0], false, false);
            acc[m][1] = __builtin_amdgcn_wmma_f32_16x16x32_bf16(
                false, a, false, b1, (short)0, acc[m][1], false, false);
        }

        if (more) {
            wait_async0();
            __syncthreads();
        }
    }

    const int gc0 = col0 + wn * 32 + l15;
    const int gc1 = gc0 + 16;
    const float bv0 = bias[gc0];
    const float bv1 = bias[gc1];
#pragma unroll
    for (int m = 0; m < 4; ++m) {
#pragma unroll
        for (int r = 0; r < 8; ++r) {
            const int gr = row0 + wm * 64 + m * 16 + r + kb;
            const float w = wgt[(size_t)gr * 8 + expert];
            const float v0 = w * (acc[m][0][r] + bv0);
            const float v1 = w * (acc[m][1][r] + bv1);
            const size_t o0 = (size_t)gr * N + gc0;
            const size_t o1 = (size_t)gr * N + gc1;
            if (accumulate) { out[o0] += v0; out[o1] += v1; }
            else            { out[o0]  = v0; out[o1]  = v1; }
        }
    }
}

// ---------------------------------------------------------------------------
// Host launcher
// ---------------------------------------------------------------------------
extern "C" void kernel_launch(void* const* d_in, const int* in_sizes, int n_in,
                              void* d_out, int out_size, void* d_ws, size_t ws_size,
                              hipStream_t stream) {
    constexpr int B = 8192, D = 1024, H = 4096, O = 1024, E = 8;

    const float* x  = (const float*)d_in[0];
    const float* Wg = (const float*)d_in[1];
    const float* W1 = (const float*)d_in[2];
    const float* b1 = (const float*)d_in[3];
    const float* W2 = (const float*)d_in[4];
    const float* b2 = (const float*)d_in[5];
    float*       out = (float*)d_out;

    // Workspace layout (bytes):
    //   wgt  : B*E*4  = 256 KB | xb : B*D*2 = 16 MB | hbuf : B*H*2 = 64 MB
    //   w1T  : H*D*2  =   8 MB | w2T: O*H*2 =  8 MB
    char* ws = (char*)d_ws;
    float*  wgt  = (float*)ws;                    ws += (size_t)B * E * sizeof(float);
    __bf16* xb   = (__bf16*)ws;                   ws += (size_t)B * D * 2;
    __bf16* hbuf = (__bf16*)ws;                   ws += (size_t)B * H * 2;
    __bf16* w1T  = (__bf16*)ws;                   ws += (size_t)H * D * 2;
    __bf16* w2T  = (__bf16*)ws;

    moe_router<<<B / 8, 256, 0, stream>>>(x, Wg, wgt, D);
    moe_cvt_x<<<(B * D) / 1024, 256, 0, stream>>>(x, xb);

    const dim3 g1(H / BN, B / BM);      // (32, 64)
    const dim3 g2(O / BN, B / BM);      // (8, 64)
    const dim3 t1(D / 64, H / 64);      // W1 transpose tiles (16, 64)
    const dim3 t2(H / 64, O / 64);      // W2 transpose tiles (64, 16)
    for (int e = 0; e < E; ++e) {
        moe_cvt_wT<<<t1, 256, 0, stream>>>(W1 + (size_t)e * D * H, w1T, D, H);
        moe_gemm1_relu<<<g1, 256, 0, stream>>>(xb, w1T, b1 + (size_t)e * H, hbuf, D, H);
        moe_cvt_wT<<<t2, 256, 0, stream>>>(W2 + (size_t)e * H * O, w2T, H, O);
        moe_gemm2_scale<<<g2, 256, 0, stream>>>(hbuf, w2T, b2 + (size_t)e * O, wgt, out,
                                                H, O, e, (e > 0) ? 1 : 0);
    }
}
